// beam_search_decoder_80796924773035
// MI455X (gfx1250) — matrix operand order repaired and checked by reading, not compiled, not run on hardware
//
#include <hip/hip_runtime.h>
#include <hip/hip_bf16.h>
#include <stdint.h>

// ---------------------------------------------------------------------------
// Beam-search decoder (K=10 beams, T=70 steps) for MI455X / gfx1250.
// Strategy: convert all big weight matrices fp32->bf16 once per launch
// (97 MB total -> L2-resident across the 70 steps), run every GEMM through
// v_wmma_f32_16x16x32_bf16 with M padded to one 16-row tile, and keep the
// sequential beam state in workspace ping-pong buffers.
// ---------------------------------------------------------------------------

#define KB   10      // beams
#define TT   70      // decode steps
#define EE   1024
#define HH   1024
#define VV   32000
#define SS   128
#define NEGF (-1.0e9f)

typedef __bf16 bf16_t;
typedef bf16_t v16bf __attribute__((ext_vector_type(16)));
typedef float  v8f   __attribute__((ext_vector_type(8)));

// ---------------- fp32 -> bf16 (round to nearest even) ----------------------
__device__ __forceinline__ unsigned short f2bf(float x) {
    union { float f; unsigned u; } v; v.f = x;
    if (x != x) return 0x7FC0;
    unsigned r = v.u + 0x7FFFu + ((v.u >> 16) & 1u);
    return (unsigned short)(r >> 16);
}

__global__ void cvt_f32_bf16(unsigned short* __restrict__ dst,
                             const float* __restrict__ src, size_t n) {
    size_t i = (size_t)blockIdx.x * blockDim.x + threadIdx.x;
    size_t stride = (size_t)gridDim.x * blockDim.x;
    for (; i < n; i += stride) dst[i] = f2bf(src[i]);
}

// ---------------- WMMA fragment loaders (per CDNA5 ISA layouts) -------------
// A (16x32 bf16, MxK): lanes 0-15 -> M=lane, K halves {0..7,16..23};
//                      lanes16-31 -> M=lane-16, K halves {8..15,24..31}
__device__ __forceinline__ v16bf load_a_frag(const bf16_t* p, int laneHi) {
    union { uint4 u[2]; v16bf v; } t;
    const int o = laneHi * 8;
    t.u[0] = *reinterpret_cast<const uint4*>(p + o);        // 8 bf16
    t.u[1] = *reinterpret_cast<const uint4*>(p + 16 + o);   // 8 bf16
    return t.v;
}
// B (32x16 bf16, KxN): lanes 0-15 -> N=lane, K {0..7,16..23}; lanes16-31 -> K {8..15,24..31}
__device__ __forceinline__ v16bf load_b_frag(const bf16_t* p, int laneHi) {
    union { uint4 u[2]; v16bf v; } t;
    const int o = laneHi * 8;
    t.u[0] = *reinterpret_cast<const uint4*>(p + o);
    t.u[1] = *reinterpret_cast<const uint4*>(p + 16 + o);
    return t.v;
}

// ---------------- Generic WMMA GEMM:  C[M,N] (+)= A[M,K] * B[N,K]^T ---------
// A: bf16, padded to Mtiles*16 rows, row stride lda (elements, mult of 32)
// B: bf16 weights, row-major [N, K], row stride ldb
// One wave per 16x16 output tile.
__global__ void wmma_gemm_bf16(const unsigned short* __restrict__ Au,
                               int lda,
                               const unsigned short* __restrict__ Bu,
                               int ldb,
                               float* __restrict__ C, int ldc,
                               int M, int Mtiles, int N, int Kdim,
                               const float* __restrict__ bias0,
                               const float* __restrict__ bias1,
                               int accum, int act_tanh) {
    const bf16_t* A = reinterpret_cast<const bf16_t*>(Au);
    const bf16_t* B = reinterpret_cast<const bf16_t*>(Bu);
    int wid  = (int)((blockIdx.x * blockDim.x + threadIdx.x) >> 5);
    int lane = (int)(threadIdx.x & 31);
    int ntiles = N >> 4;
    if (wid >= Mtiles * ntiles) return;
    int mt = wid % Mtiles;
    int nt = wid / Mtiles;
    int laneHi = lane >> 4;
    int lmod   = lane & 15;

    const bf16_t* Ap = A + (size_t)(mt * 16 + lmod) * lda;
    const bf16_t* Bp = B + (size_t)(nt * 16 + lmod) * ldb;

    v8f acc = {};
    for (int k = 0; k < Kdim; k += 32) {
        v16bf a = load_a_frag(Ap + k, laneHi);
        v16bf b = load_b_frag(Bp + k, laneHi);
        acc = __builtin_amdgcn_wmma_f32_16x16x32_bf16(
            false, a, false, b, (short)0, acc, false, false);
    }

    int col = nt * 16 + lmod;
#pragma unroll
    for (int r = 0; r < 8; ++r) {
        int m = mt * 16 + r + laneHi * 8;   // C layout: VGPR r -> M=r / r+8
        if (m < M) {
            float v = acc[r];
            if (accum)  v += C[(size_t)m * ldc + col];
            if (bias0)  v += bias0[col];
            if (bias1)  v += bias1[col];
            if (act_tanh) v = tanhf(v);
            C[(size_t)m * ldc + col] = v;
        }
    }
}

// ---------------- packing kernels (fp32 state -> padded bf16 A tiles) -------
__global__ void pack_x_kernel(unsigned short* __restrict__ x_bf,   // [16][2048]
                              const float* __restrict__ emb,       // [V][1024]
                              const float* __restrict__ prevOut,   // [10][1024]
                              const int* __restrict__ tokens_cur,  // [10][71]
                              int t) {
    int i = blockIdx.x * blockDim.x + threadIdx.x;
    if (i >= 16 * 2048) return;
    int m = i >> 11, col = i & 2047;
    unsigned short o = 0;
    if (m < KB) {
        if (col < 1024) {
            int tok = tokens_cur[m * (TT + 1) + t];
            o = f2bf(emb[(size_t)tok * 1024 + col]);
        } else {
            o = f2bf(prevOut[m * 1024 + (col - 1024)]);
        }
    }
    x_bf[i] = o;
}

__global__ void pack_rows_kernel(unsigned short* __restrict__ dst, // [16][w]
                                 const float* __restrict__ src,    // [10][w]
                                 int w) {
    int i = blockIdx.x * blockDim.x + threadIdx.x;
    if (i >= 16 * w) return;
    int m = i / w, col = i % w;
    dst[i] = (m < KB) ? f2bf(src[m * w + col]) : (unsigned short)0;
}

__global__ void pack_y_kernel(unsigned short* __restrict__ y_bf,   // [16][3072]
                              const float* __restrict__ ctx,       // [10][2048]
                              const float* __restrict__ h_new) {   // [10][1024]
    int i = blockIdx.x * blockDim.x + threadIdx.x;
    if (i >= 16 * 3072) return;
    int m = i / 3072, col = i % 3072;
    unsigned short o = 0;
    if (m < KB)
        o = (col < 2048) ? f2bf(ctx[m * 2048 + col])
                         : f2bf(h_new[m * 1024 + (col - 2048)]);
    y_bf[i] = o;
}

// ---------------- LSTM cell elementwise -------------------------------------
__device__ __forceinline__ float sigmoidf_(float x) { return 1.f / (1.f + __expf(-x)); }

__global__ void lstm_kernel(const float* __restrict__ gates,  // [10(16)][4096]
                            const float* __restrict__ c_old,  // [10][1024]
                            float* __restrict__ c_new,
                            float* __restrict__ h_new) {
    int i = blockIdx.x * blockDim.x + threadIdx.x;
    if (i >= KB * HH) return;
    int k = i >> 10, j = i & 1023;
    const float* g = gates + (size_t)k * 4096;
    float gi = g[j], gf = g[1024 + j], gg = g[2048 + j], go = g[3072 + j];
    float cn = sigmoidf_(gf) * c_old[i] + sigmoidf_(gi) * tanhf(gg);
    c_new[i] = cn;
    h_new[i] = sigmoidf_(go) * tanhf(cn);
}

// ---------------- attention ------------------------------------------------
__global__ void attn_scores_kernel(const float* __restrict__ keys,  // [128][1024]
                                   const float* __restrict__ h_new, // [10][1024]
                                   const int* __restrict__ src_len,
                                   float* __restrict__ e) {         // [10][128]
    int wid = (int)((blockIdx.x * blockDim.x + threadIdx.x) >> 5);
    if (wid >= KB * SS) return;
    int lane = (int)(threadIdx.x & 31);
    int k = wid >> 7, s = wid & 127;
    const float* kr = keys + (size_t)s * 1024;
    const float* hr = h_new + (size_t)k * 1024;
    float sum = 0.f;
    for (int i = lane; i < 1024; i += 32) sum += kr[i] * hr[i];
    for (int off = 16; off; off >>= 1) sum += __shfl_xor(sum, off, 32);
    if (lane == 0) e[k * SS + s] = (s >= src_len[0]) ? NEGF : sum;
}

__global__ void softmax128_kernel(float* __restrict__ e) {  // in-place per beam
    __shared__ float red[SS];
    int k = blockIdx.x, s = threadIdx.x;
    float v = e[k * SS + s];
    red[s] = v; __syncthreads();
    for (int off = 64; off; off >>= 1) {
        if (s < off) red[s] = fmaxf(red[s], red[s + off]);
        __syncthreads();
    }
    float mx = red[0]; __syncthreads();
    float ex = __expf(v - mx);
    red[s] = ex; __syncthreads();
    for (int off = 64; off; off >>= 1) {
        if (s < off) red[s] += red[s + off];
        __syncthreads();
    }
    e[k * SS + s] = ex / red[0];
}

__global__ void ctx_kernel(const float* __restrict__ alpha,  // [10][128]
                           const float* __restrict__ enc,    // [128][2048]
                           float* __restrict__ ctx) {        // [10][2048]
    int i = blockIdx.x * blockDim.x + threadIdx.x;
    if (i >= KB * 2048) return;
    int k = i >> 11, v = i & 2047;
    float s = 0.f;
    for (int si = 0; si < SS; ++si)
        s += alpha[k * SS + si] * enc[(size_t)si * 2048 + v];
    ctx[i] = s;
}

// ---------------- log-softmax stats + candidates ---------------------------
__global__ void rowstat_kernel(const float* __restrict__ logits, // [10][32000]
                               float* __restrict__ rstat) {      // [10][2]
    __shared__ float red[256];
    int k = blockIdx.x, tid = threadIdx.x;
    const float* row = logits + (size_t)k * VV;
    float mx = -3.4e38f;
    for (int v = tid; v < VV; v += 256) mx = fmaxf(mx, row[v]);
    red[tid] = mx; __syncthreads();
    for (int off = 128; off; off >>= 1) {
        if (tid < off) red[tid] = fmaxf(red[tid], red[tid + off]);
        __syncthreads();
    }
    mx = red[0]; __syncthreads();
    float sm = 0.f;
    for (int v = tid; v < VV; v += 256) sm += __expf(row[v] - mx);
    red[tid] = sm; __syncthreads();
    for (int off = 128; off; off >>= 1) {
        if (tid < off) red[tid] += red[tid + off];
        __syncthreads();
    }
    if (tid == 0) { rstat[2 * k] = mx; rstat[2 * k + 1] = logf(red[0]); }
}

__global__ void cand_kernel(const float* __restrict__ logits,
                            const float* __restrict__ rstat,
                            const float* __restrict__ scores,
                            const int* __restrict__ finished,
                            float* __restrict__ cand) {      // [320000]
    int i = blockIdx.x * blockDim.x + threadIdx.x;
    if (i >= KB * VV) return;
    int k = i / VV, v = i - k * VV;
    float c;
    if (finished[k]) c = scores[k] + ((v == 0) ? 0.f : NEGF);  // PAD = 0
    else             c = scores[k] + (logits[i] - rstat[2 * k] - rstat[2 * k + 1]);
    cand[i] = c;
}

// ---------------- top-10 (10 sequential block argmax passes) ----------------
__global__ void topk_kernel(float* __restrict__ cand,
                            float* __restrict__ topv,
                            int* __restrict__ topi) {
    __shared__ float sv[512];
    __shared__ int   si[512];
    int tid = threadIdx.x;
    for (int sel = 0; sel < KB; ++sel) {
        float bv = -3.4e38f; int bi = 0x7fffffff;
        for (int i = tid; i < KB * VV; i += 512) {
            float v = cand[i];
            if (v > bv || (v == bv && i < bi)) { bv = v; bi = i; }
        }
        sv[tid] = bv; si[tid] = bi; __syncthreads();
        for (int off = 256; off; off >>= 1) {
            if (tid < off) {
                if (sv[tid + off] > sv[tid] ||
                    (sv[tid + off] == sv[tid] && si[tid + off] < si[tid])) {
                    sv[tid] = sv[tid + off]; si[tid] = si[tid + off];
                }
            }
            __syncthreads();
        }
        if (tid == 0) {
            topv[sel] = sv[0]; topi[sel] = si[0];
            cand[si[0]] = -3.0e38f;           // exclude from next pass
        }
        __syncthreads();
    }
}

// ---------------- beam reorder / state update -------------------------------
__global__ void update_kernel(const int* __restrict__ topi,
                              const float* __restrict__ topv,
                              const int* __restrict__ tokens_cur,
                              int* __restrict__ tokens_nxt,
                              float* __restrict__ scores_nxt,
                              const int* __restrict__ fin_cur,
                              int* __restrict__ fin_nxt,
                              const float* __restrict__ h_new,
                              const float* __restrict__ c_new,
                              const float* __restrict__ outb,
                              float* __restrict__ h_state,
                              float* __restrict__ c_state,
                              float* __restrict__ prev_state,
                              int t) {
    int kn = blockIdx.x, tid = threadIdx.x;
    int idx = topi[kn];
    int row = idx / VV;
    int col = idx - row * VV;
    for (int j = tid; j < HH; j += 256) {
        h_state[kn * HH + j]    = h_new[row * HH + j];
        c_state[kn * HH + j]    = c_new[row * HH + j];
        prev_state[kn * HH + j] = outb[row * HH + j];
    }
    for (int j = tid; j <= TT; j += 256)
        tokens_nxt[kn * (TT + 1) + j] = tokens_cur[row * (TT + 1) + j];
    __syncthreads();
    if (tid == 0) {
        tokens_nxt[kn * (TT + 1) + t + 1] = col;
        scores_nxt[kn] = topv[kn];
        fin_nxt[kn] = fin_cur[row] | (col == 2 ? 1 : 0);    // EOS = 2
    }
}

// ---------------- init & final output ---------------------------------------
__global__ void init_kernel(float* __restrict__ h_state,
                            float* __restrict__ c_state,
                            float* __restrict__ prev_state,
                            const float* __restrict__ h0,
                            const float* __restrict__ c0,
                            int* __restrict__ tokens0,
                            float* __restrict__ scores0,
                            int* __restrict__ fin0) {
    int i = blockIdx.x * blockDim.x + threadIdx.x;
    if (i < KB * HH) {
        int j = i & 1023;
        h_state[i] = h0[j];
        c_state[i] = c0[j];
        prev_state[i] = 0.f;
    }
    if (i < KB * (TT + 1)) tokens0[i] = ((i % (TT + 1)) == 0) ? 1 : 0;  // START/PAD
    if (i < KB) { scores0[i] = (i == 0) ? 0.f : NEGF; fin0[i] = 0; }
}

__global__ void final_kernel(const int* __restrict__ tokens,
                             const float* __restrict__ scores,
                             float* __restrict__ out) {
    int i = blockIdx.x * blockDim.x + threadIdx.x;
    int nt = KB * (TT + 1);
    if (i < nt) out[i] = (float)tokens[i];
    else if (i < nt + KB) out[i] = scores[i - nt];
}

// ---------------------------------------------------------------------------
static inline int ceil_div(int a, int b) { return (a + b - 1) / b; }

extern "C" void kernel_launch(void* const* d_in, const int* in_sizes, int n_in,
                              void* d_out, int out_size, void* d_ws, size_t ws_size,
                              hipStream_t stream) {
    const float* enc      = (const float*)d_in[0];   // [128][2048]
    const float* h0       = (const float*)d_in[1];   // [1024]
    const float* c0       = (const float*)d_in[2];   // [1024]
    const float* emb      = (const float*)d_in[3];   // [32000][1024]
    const float* W_ih     = (const float*)d_in[4];   // [4096][2048]
    const float* b_ih     = (const float*)d_in[5];   // [4096]
    const float* W_hh     = (const float*)d_in[6];   // [4096][1024]
    const float* b_hh     = (const float*)d_in[7];   // [4096]
    const float* W_attprj = (const float*)d_in[8];   // [1024][2048]
    const float* W_comb   = (const float*)d_in[9];   // [1024][3072]
    const float* W_vocab  = (const float*)d_in[10];  // [32000][1024]
    const int*   src_len  = (const int*)d_in[11];
    float* out = (float*)d_out;

    // ---- workspace carve-up (all 256B aligned) ----
    char* base = (char*)d_ws;
    size_t off = 0;
    auto take = [&](size_t bytes) -> char* {
        char* p = base + off;
        off += (bytes + 255) & ~(size_t)255;
        return p;
    };
    unsigned short* Wih_bf   = (unsigned short*)take((size_t)4096 * 2048 * 2);
    unsigned short* Whh_bf   = (unsigned short*)take((size_t)4096 * 1024 * 2);
    unsigned short* Wcomb_bf = (unsigned short*)take((size_t)1024 * 3072 * 2);
    unsigned short* Wvoc_bf  = (unsigned short*)take((size_t)32000 * 1024 * 2);
    unsigned short* Watt_bf  = (unsigned short*)take((size_t)1024 * 2048 * 2);
    unsigned short* enc_bf   = (unsigned short*)take((size_t)128 * 2048 * 2);
    float* keys    = (float*)take((size_t)128 * 1024 * 4);
    unsigned short* x_bf = (unsigned short*)take((size_t)16 * 2048 * 2);
    unsigned short* h_bf = (unsigned short*)take((size_t)16 * 1024 * 2);
    unsigned short* y_bf = (unsigned short*)take((size_t)16 * 3072 * 2);
    unsigned short* o_bf = (unsigned short*)take((size_t)16 * 1024 * 2);
    float* gates   = (float*)take((size_t)16 * 4096 * 4);
    float* h_state = (float*)take((size_t)KB * HH * 4);
    float* c_state = (float*)take((size_t)KB * HH * 4);
    float* p_state = (float*)take((size_t)KB * HH * 4);
    float* h_new   = (float*)take((size_t)KB * HH * 4);
    float* c_new   = (float*)take((size_t)KB * HH * 4);
    float* ebuf    = (float*)take((size_t)KB * SS * 4);
    float* ctx     = (float*)take((size_t)KB * 2048 * 4);
    float* outb    = (float*)take((size_t)KB * HH * 4);
    float* logits  = (float*)take((size_t)KB * VV * 4);
    float* rstat   = (float*)take((size_t)KB * 2 * 4);
    float* cand    = (float*)take((size_t)KB * VV * 4);
    float* topv    = (float*)take(64);
    int*   topi    = (int*)take(64);
    int*   tokens2 = (int*)take((size_t)2 * KB * (TT + 1) * 4);
    float* scores2 = (float*)take((size_t)2 * KB * 4);
    int*   fin2    = (int*)take((size_t)2 * KB * 4);
    (void)ws_size; // requires ~106 MB of workspace

    // ---- once-per-launch: weight conversions + attention keys ----
    cvt_f32_bf16<<<2048, 256, 0, stream>>>(Wih_bf,   W_ih,     (size_t)4096 * 2048);
    cvt_f32_bf16<<<2048, 256, 0, stream>>>(Whh_bf,   W_hh,     (size_t)4096 * 1024);
    cvt_f32_bf16<<<2048, 256, 0, stream>>>(Wcomb_bf, W_comb,   (size_t)1024 * 3072);
    cvt_f32_bf16<<<4096, 256, 0, stream>>>(Wvoc_bf,  W_vocab,  (size_t)32000 * 1024);
    cvt_f32_bf16<<<1024, 256, 0, stream>>>(Watt_bf,  W_attprj, (size_t)1024 * 2048);
    cvt_f32_bf16<<<256,  256, 0, stream>>>(enc_bf,   enc,      (size_t)128 * 2048);

    // keys[128][1024] = enc[128][2048] @ W_attproj^T  -> 8 M-tiles x 64 N-tiles
    wmma_gemm_bf16<<<ceil_div(8 * 64 * 32, 256), 256, 0, stream>>>(
        enc_bf, 2048, Watt_bf, 2048, keys, 1024,
        128, 8, 1024, 2048, nullptr, nullptr, 0, 0);

    init_kernel<<<ceil_div(KB * HH, 256), 256, 0, stream>>>(
        h_state, c_state, p_state, h0, c0, tokens2, scores2, fin2);

    // ---- sequential decode loop ----
    for (int t = 0; t < TT; ++t) {
        int cur = t & 1, nxt = cur ^ 1;
        int*   tok_c = tokens2 + cur * KB * (TT + 1);
        int*   tok_n = tokens2 + nxt * KB * (TT + 1);
        float* sc_c  = scores2 + cur * KB;
        float* sc_n  = scores2 + nxt * KB;
        int*   fi_c  = fin2 + cur * KB;
        int*   fi_n  = fin2 + nxt * KB;

        pack_x_kernel<<<ceil_div(16 * 2048, 256), 256, 0, stream>>>(
            x_bf, emb, p_state, tok_c, t);
        pack_rows_kernel<<<ceil_div(16 * 1024, 256), 256, 0, stream>>>(
            h_bf, h_state, 1024);

        // gates = x @ W_ih^T  ;  gates += h @ W_hh^T + b_ih + b_hh
        wmma_gemm_bf16<<<32, 256, 0, stream>>>(
            x_bf, 2048, Wih_bf, 2048, gates, 4096,
            KB, 1, 4096, 2048, nullptr, nullptr, 0, 0);
        wmma_gemm_bf16<<<32, 256, 0, stream>>>(
            h_bf, 1024, Whh_bf, 1024, gates, 4096,
            KB, 1, 4096, 1024, b_ih, b_hh, 1, 0);

        lstm_kernel<<<ceil_div(KB * HH, 256), 256, 0, stream>>>(
            gates, c_state, c_new, h_new);

        attn_scores_kernel<<<ceil_div(KB * SS * 32, 256), 256, 0, stream>>>(
            keys, h_new, src_len, ebuf);
        softmax128_kernel<<<KB, SS, 0, stream>>>(ebuf);
        ctx_kernel<<<ceil_div(KB * 2048, 256), 256, 0, stream>>>(ebuf, enc, ctx);

        pack_y_kernel<<<ceil_div(16 * 3072, 256), 256, 0, stream>>>(y_bf, ctx, h_new);
        // out = tanh(concat(ctx,h) @ W_comb^T)
        wmma_gemm_bf16<<<8, 256, 0, stream>>>(
            y_bf, 3072, Wcomb_bf, 3072, outb, 1024,
            KB, 1, 1024, 3072, nullptr, nullptr, 0, 1);

        pack_rows_kernel<<<ceil_div(16 * 1024, 256), 256, 0, stream>>>(o_bf, outb, 1024);
        // logits = out @ W_vocab^T  (the big streamed GEMM: 65 MB bf16, L2-hot)
        wmma_gemm_bf16<<<250, 256, 0, stream>>>(
            o_bf, 1024, Wvoc_bf, 1024, logits, 32000,
            KB, 1, 32000, 1024, nullptr, nullptr, 0, 0);

        rowstat_kernel<<<KB, 256, 0, stream>>>(logits, rstat);
        cand_kernel<<<ceil_div(KB * VV, 256), 256, 0, stream>>>(
            logits, rstat, sc_c, fi_c, cand);
        topk_kernel<<<1, 512, 0, stream>>>(cand, topv, topi);

        update_kernel<<<KB, 256, 0, stream>>>(
            topi, topv, tok_c, tok_n, sc_n, fi_c, fi_n,
            h_new, c_new, outb, h_state, c_state, p_state, t);
    }

    // after 70 steps the live buffers are index (70 & 1) = 0
    final_kernel<<<ceil_div(KB * (TT + 1) + KB, 256), 256, 0, stream>>>(
        tokens2, scores2, out);
    (void)in_sizes; (void)n_in; (void)out_size;
}
